// QNNQwen3Attention_1486058684891
// MI455X (gfx1250) — matrix operand order repaired and checked by reading, not compiled, not run on hardware
//
#include <hip/hip_runtime.h>
#include <hip/hip_bf16.h>
#include <math.h>

// Problem constants (from reference setup_inputs)
#define H   2048
#define HQ  16
#define HKV 8
#define DH  128
#define CC  16384            // cached positions; new token handled in reduce

// Workspace layout (float offsets)
#define WQOFF   0            // q after norm+rope           [16*128]
#define WKOFF   2048         // k_new after norm+rope       [8*128]
#define WVOFF   3072         // v_new                       [8*128]
#define AOOFF   4096         // attention output, head-major[16*128]
#define PARTOFF 6144         // 16 qheads * 64 partials * (m,l,acc[128])
#define PSTRIDE 130
#define NPART   64           // partials per q head (8 chunks * 8 waves)
#define TPW     16           // tiles (of 16 positions) per wave: 64*16*16 = 16384

typedef __attribute__((ext_vector_type(16))) __bf16 v16bf;
typedef __attribute__((ext_vector_type(8)))  float  v8f;

union BfFrag { v16bf v; };

// ---------------------------------------------------------------------------
// Kernel 1: per-head QKV projection + RMSNorm + RoPE.  One block per head.
//   heads 0..15 -> q, 16..23 -> k, 24..31 -> v
// ---------------------------------------------------------------------------
__global__ void __launch_bounds__(256) qkv_kernel(
    const float* __restrict__ x_in, const float* __restrict__ cosv,
    const float* __restrict__ sinv,
    const float* __restrict__ Wq, const float* __restrict__ Wk,
    const float* __restrict__ Wv,
    const float* __restrict__ qw, const float* __restrict__ kw,
    float* __restrict__ ws, float* __restrict__ out)
{
    __shared__ float xs[H];
    __shared__ float hv[DH];
    __shared__ float nrm[DH];
    __shared__ float ssum;
    const int tid  = threadIdx.x;
    const int head = blockIdx.x;
    for (int i = tid; i < H; i += 256) xs[i] = x_in[i];
    __syncthreads();

    const int lane = tid & 31, wv = tid >> 5;
    const float* Wbase;
    if (head < HQ)            Wbase = Wq + (size_t)head        * DH * H;
    else if (head < HQ + HKV) Wbase = Wk + (size_t)(head - HQ) * DH * H;
    else                      Wbase = Wv + (size_t)(head - HQ - HKV) * DH * H;

    // wave-per-row matvec: 8 waves cover 128 rows
    for (int r = wv; r < DH; r += 8) {
        const float4* wr = (const float4*)(Wbase + (size_t)r * H);
        const float4* xr = (const float4*)xs;
        float acc = 0.f;
        for (int i = lane; i < H / 4; i += 32) {
            float4 a = wr[i], b = xr[i];
            acc = fmaf(a.x, b.x, acc); acc = fmaf(a.y, b.y, acc);
            acc = fmaf(a.z, b.z, acc); acc = fmaf(a.w, b.w, acc);
        }
        for (int off = 16; off; off >>= 1) acc += __shfl_xor(acc, off);
        if (lane == 0) hv[r] = acc;
    }
    __syncthreads();

    if (head < HQ + HKV) {
        if (tid == 0) {
            float s = 0.f;
            for (int d = 0; d < DH; ++d) s += hv[d] * hv[d];
            ssum = s;
        }
        __syncthreads();
        const float* nw = (head < HQ) ? qw : kw;
        float inv = rsqrtf(ssum * (1.0f / DH) + 1e-6f);
        if (tid < DH) nrm[tid] = hv[tid] * inv * nw[tid];
        __syncthreads();
        if (tid < DH) {
            int d = tid;
            float rot = (d < DH / 2) ? -nrm[d + DH / 2] : nrm[d - DH / 2];
            float val = nrm[d] * cosv[d] + rot * sinv[d];
            if (head < HQ) {
                ws[WQOFF + head * DH + d] = val;
            } else {
                int kh = head - HQ;
                ws[WKOFF + kh * DH + d] = val;
                out[2048 + kh * DH + d] = val;        // k_new output
            }
        }
    } else {
        if (tid < DH) {
            int vh2 = head - HQ - HKV;
            float val = hv[tid];
            ws[WVOFF + vh2 * DH + tid] = val;
            out[3072 + vh2 * DH + tid] = val;          // v output
        }
    }
}

// ---------------------------------------------------------------------------
// Kernel 2: flash-decode partials over the 16384 cached positions.
// grid = (8 chunks, 8 kv heads) x 256 threads.  Each wave owns exactly 16
// tiles of 16 positions; scores via chained v_wmma_f32_16x16x32_bf16.
// Fully uniform: no branches in the tile loop; B-fragment loads are
// base + e*CC immediate offsets; mask load is unconditional + cndmask.
// ---------------------------------------------------------------------------
__global__ void __launch_bounds__(256) attn_partial_kernel(
    const float* __restrict__ kc, const float* __restrict__ vc,
    const float* __restrict__ mask, float* __restrict__ ws)
{
    const int h     = blockIdx.y;
    const int chunk = blockIdx.x;
    const int tid   = threadIdx.x;
    const int lane  = tid & 31, wvid = tid >> 5;
    const int part  = chunk * 8 + wvid;              // 0..63
    const int t0    = part * TPW;
    const int col   = lane & 15;                     // N index / M row
    const int half  = lane >> 4;
    const float NEG = -3e38f;

    // A fragments: rows 0,1 = the two q heads of kv head h; rows 2..15 zero.
    // A k order used here (contiguous, same as B): K(e,lane) = e + 16*(lane>>4)
    BfFrag A[4];
    for (int kb = 0; kb < 4; ++kb) {
        #pragma unroll
        for (int e = 0; e < 16; ++e) {
            int K = e + (half << 4);
            float f = (col < 2) ? ws[WQOFF + (2 * h + col) * DH + kb * 32 + K] : 0.f;
            A[kb].v[e] = (__bf16)f;
        }
    }

    float m0 = NEG, l0 = 0.f, l1 = 0.f;
    float acc0[4] = {0, 0, 0, 0}, acc1[4] = {0, 0, 0, 0};
    const float scale = 0.08838834764831845f;        // 128^-0.5
    const float* kh = kc + (size_t)h * DH * CC;      // [DH][CC]
    const float* vh = vc + (size_t)h * CC * DH;      // [CC][DH]

    for (int t = t0; t < t0 + TPW; ++t) {
        const int c = t * 16 + col;
        // one base address per kb; 16 element loads at immediate offsets e*CC
        const float* kcol = kh + (size_t)(half << 4) * CC + c;
        __builtin_prefetch(kcol + 16, 0, 1);          // next tile (global_prefetch_b8)
        const float mk = mask[c];                     // unconditional, c < CC all lanes

        v8f cacc = {};
        #pragma unroll
        for (int kb = 0; kb < 4; ++kb) {
            // B k order used here (contiguous, same as A): K(e,lane) = e + 16*(lane>>4), N = lane&15
            BfFrag B;
            const float* kb_base = kcol + (size_t)(kb * 32) * CC;
            #pragma unroll
            for (int e = 0; e < 16; ++e)
                B.v[e] = (__bf16)kb_base[(size_t)e * CC];
            cacc = __builtin_amdgcn_wmma_f32_16x16x32_bf16(
                false, A[kb].v, false, B.v, (short)0, cacc, false, false);
        }

        // scores for the two q heads live in C rows M=0 (cacc[0]) and M=1 (cacc[1]);
        // lanes 16..31 are parked at NEG via cndmask (no branch)
        float s0 = (half == 0) ? fmaf(cacc[0], scale, mk) : NEG;
        float s1 = (half == 0) ? fmaf(cacc[1], scale, mk) : NEG;

        float tmax = fmaxf(s0, s1);
        for (int off = 16; off; off >>= 1) tmax = fmaxf(tmax, __shfl_xor(tmax, off));

        float mn   = fmaxf(m0, tmax);
        float corr = __expf(m0 - mn);
        l0 *= corr; l1 *= corr;
        #pragma unroll
        for (int j = 0; j < 4; ++j) { acc0[j] *= corr; acc1[j] *= corr; }
        float e0 = __expf(s0 - mn);                  // 0 on lanes 16..31
        float e1 = __expf(s1 - mn);

        const float* vrow = vh + (size_t)(t * 16) * DH + lane * 4;
        #pragma unroll
        for (int n = 0; n < 16; ++n) {
            float p0 = __shfl(e0, n);
            float p1 = __shfl(e1, n);
            float4 vv = *(const float4*)(vrow + (size_t)n * DH);
            l0 += p0; l1 += p1;
            acc0[0] = fmaf(p0, vv.x, acc0[0]); acc0[1] = fmaf(p0, vv.y, acc0[1]);
            acc0[2] = fmaf(p0, vv.z, acc0[2]); acc0[3] = fmaf(p0, vv.w, acc0[3]);
            acc1[0] = fmaf(p1, vv.x, acc1[0]); acc1[1] = fmaf(p1, vv.y, acc1[1]);
            acc1[2] = fmaf(p1, vv.z, acc1[2]); acc1[3] = fmaf(p1, vv.w, acc1[3]);
        }
        m0 = mn;
    }

    float* P0 = ws + PARTOFF + (size_t)((2 * h)     * NPART + part) * PSTRIDE;
    float* P1 = ws + PARTOFF + (size_t)((2 * h + 1) * NPART + part) * PSTRIDE;
    if (lane == 0) { P0[0] = m0; P0[1] = l0; P1[0] = m0; P1[1] = l1; }
    #pragma unroll
    for (int j = 0; j < 4; ++j) {
        P0[2 + lane * 4 + j] = acc0[j];
        P1[2 + lane * 4 + j] = acc1[j];
    }
}

// ---------------------------------------------------------------------------
// Kernel 3: merge 64 cache partials per q head + the new-token contribution
// (score = q . k_new, value = v_new) as one analytic partial {m=s,l=1,acc=v}.
// 16 blocks x 128 threads.
// ---------------------------------------------------------------------------
__global__ void __launch_bounds__(128) attn_reduce_kernel(
    const float* __restrict__ mask, float* __restrict__ ws)
{
    __shared__ float red[DH];
    const int qh  = blockIdx.x;
    const int d   = threadIdx.x;
    const int kh2 = qh >> 1;
    const float scale = 0.08838834764831845f;

    // new-token score via LDS tree reduce
    red[d] = ws[WQOFF + qh * DH + d] * ws[WKOFF + kh2 * DH + d];
    __syncthreads();
    for (int s = 64; s; s >>= 1) {
        if (d < s) red[d] += red[d + s];
        __syncthreads();
    }
    const float snew = red[0] * scale + mask[CC];

    const float* base = ws + PARTOFF + (size_t)qh * NPART * PSTRIDE;
    float M = snew;
    for (int p = 0; p < NPART; ++p) M = fmaxf(M, base[p * PSTRIDE]);
    float wn = __expf(snew - M);
    float lt = wn;
    float at = ws[WVOFF + kh2 * DH + d] * wn;
    for (int p = 0; p < NPART; ++p) {
        float w = __expf(base[p * PSTRIDE] - M);
        lt = fmaf(base[p * PSTRIDE + 1], w, lt);
        at = fmaf(base[p * PSTRIDE + 2 + d], w, at);
    }
    ws[AOOFF + qh * DH + d] = at / lt;               // head-major, matches reference
}

// ---------------------------------------------------------------------------
// Kernel 4: o_proj matvec  attn[i] = sum_j out[j] * Wo[i][j]. 256 blk x 256.
// ---------------------------------------------------------------------------
__global__ void __launch_bounds__(256) oproj_kernel(
    const float* __restrict__ Wo, const float* __restrict__ ws,
    float* __restrict__ out)
{
    __shared__ float a[H];
    const int tid = threadIdx.x, lane = tid & 31, wvid = tid >> 5;
    for (int i = tid; i < H; i += 256) a[i] = ws[AOOFF + i];
    __syncthreads();
    const int r = blockIdx.x * 8 + wvid;
    const float4* wr = (const float4*)(Wo + (size_t)r * H);
    const float4* ar = (const float4*)a;
    float acc = 0.f;
    for (int i = lane; i < H / 4; i += 32) {
        float4 x = wr[i], y = ar[i];
        acc = fmaf(x.x, y.x, acc); acc = fmaf(x.y, y.y, acc);
        acc = fmaf(x.z, y.z, acc); acc = fmaf(x.w, y.w, acc);
    }
    for (int off = 16; off; off >>= 1) acc += __shfl_xor(acc, off);
    if (lane == 0) out[r] = acc;
}

// ---------------------------------------------------------------------------
extern "C" void kernel_launch(void* const* d_in, const int* in_sizes, int n_in,
                              void* d_out, int out_size, void* d_ws, size_t ws_size,
                              hipStream_t stream)
{
    const float* x    = (const float*)d_in[0];
    const float* cosv = (const float*)d_in[1];
    const float* sinv = (const float*)d_in[2];
    const float* mask = (const float*)d_in[3];
    const float* kc   = (const float*)d_in[4];
    const float* vc   = (const float*)d_in[5];
    const float* Wq   = (const float*)d_in[6];
    const float* Wk   = (const float*)d_in[7];
    const float* Wv   = (const float*)d_in[8];
    const float* Wo   = (const float*)d_in[9];
    const float* qw   = (const float*)d_in[10];
    const float* kw   = (const float*)d_in[11];
    float* out = (float*)d_out;
    float* ws  = (float*)d_ws;
    (void)in_sizes; (void)n_in; (void)out_size; (void)ws_size;

    qkv_kernel<<<32, 256, 0, stream>>>(x, cosv, sinv, Wq, Wk, Wv, qw, kw, ws, out);
    attn_partial_kernel<<<dim3(8, 8), 256, 0, stream>>>(kc, vc, mask, ws);
    attn_reduce_kernel<<<16, 128, 0, stream>>>(mask, ws);
    oproj_kernel<<<256, 256, 0, stream>>>(Wo, ws, out);
}
